// CMA_CTEBlock_35407710388554
// MI455X (gfx1250) — compile-verified, matrix operand order checked by reading, not run on hardware
//
#include <hip/hip_runtime.h>
#include <hip/hip_bf16.h>
#include <math.h>

typedef __attribute__((ext_vector_type(16))) _Float16 v16h;
typedef __attribute__((ext_vector_type(8)))  _Float16 v8h;
typedef __attribute__((ext_vector_type(8)))  float    v8f;

#define TB 256
constexpr int BB = 32, NT = 64, NS = 256, NN = 320;
constexpr int DIM = 768, HEADS = 12, HD = 64, MATCH = 64, VDIM = 64, MLPH = 3072;
constexpr float SCL = 0.125f;     // HD^-0.5
constexpr float MSCL = 0.125f;    // MATCH^-0.5
constexpr float NEGV = -1.0e9f;

// ---------------- dynamic output offsets (lens_keep read on device) ----------
struct Off {
  int LK, NKEEP;
  size_t x0, x1, gt, ki, ri, cr, ct;
};
__device__ inline Off mkoff(const int* kr) {
  Off o;
  int r = kr ? *kr : 70;
  o.LK = (r * NS + 99) / 100;          // ceil(r/100*NS)
  o.NKEEP = NT + o.LK;
  size_t xsz = (size_t)BB * o.NKEEP * DIM;
  o.x0 = 0;
  o.x1 = xsz;
  o.gt = 2 * xsz;
  o.ki = o.gt + (size_t)BB * NT;
  o.ri = o.ki + (size_t)BB * o.LK;
  o.cr = o.ri + (size_t)BB * (NS - o.LK);
  o.ct = o.cr + (size_t)BB * HEADS * NN * NN;
  return o;
}

// ---------------- WMMA helpers ----------------------------------------------
__device__ inline v8f wmma16(v16h a, v16h b, v8f c) {
  return __builtin_amdgcn_wmma_f32_16x16x32_f16(false, a, false, b, (short)0, c,
                                                false, false);
}

__device__ inline v16h cat16(v8h lo, v8h hi) {
  return __builtin_shufflevector(lo, hi, 0, 1, 2, 3, 4, 5, 6, 7,
                                 8, 9, 10, 11, 12, 13, 14, 15);
}

// A-style fragment (16x32 f16) from a row-major matrix (two b128 loads/lane).
// ISA layout: lanes 0-15 hold K=e (e<8) / K=16+e; lanes 16-31 hold K=8+e / 24+e.
__device__ inline v16h ldfrag_rm(const _Float16* m, int row0, int k0, int ld) {
  int lane = threadIdx.x & 31;
  const _Float16* p = m + (size_t)(row0 + (lane & 15)) * ld + k0 + ((lane >> 4) << 3);
  v8h lo = *(const v8h*)p;
  v8h hi = *(const v8h*)(p + 16);
  return cat16(lo, hi);
}

// Fragment from an LDS tile stored [X][32] (n-major or m-major), contiguous.
__device__ inline v16h ldfrag_lds32(const _Float16* rowbase, int koff) {
  v8h lo = *(const v8h*)(rowbase + koff);
  v8h hi = *(const v8h*)(rowbase + 16 + koff);
  return cat16(lo, hi);
}

// ---------------- misc small kernels -----------------------------------------
__global__ void k_f2h(const float* s, _Float16* d, int n) {
  int i = blockIdx.x * TB + threadIdx.x;
  if (i < n) d[i] = (_Float16)s[i];
}

__global__ void __launch_bounds__(TB) k_ln(const float* x, const float* gg,
                                           const float* bb, _Float16* out,
                                           int dyn, const int* kr) {
  int row = blockIdx.x;
  if (dyn) { Off o = mkoff(kr); if (row >= BB * o.NKEEP) return; }
  const float* xr = x + (size_t)row * DIM;
  int tid = threadIdx.x;
  float v[3], s = 0.f, s2 = 0.f;
#pragma unroll
  for (int i = 0; i < 3; i++) { v[i] = xr[tid + i * TB]; s += v[i]; s2 += v[i] * v[i]; }
  __shared__ float sh[TB];
  sh[tid] = s; __syncthreads();
  for (int st = 128; st > 0; st >>= 1) { if (tid < st) sh[tid] += sh[tid + st]; __syncthreads(); }
  float mean = sh[0] * (1.f / DIM); __syncthreads();
  sh[tid] = s2; __syncthreads();
  for (int st = 128; st > 0; st >>= 1) { if (tid < st) sh[tid] += sh[tid + st]; __syncthreads(); }
  float var = sh[0] * (1.f / DIM) - mean * mean;
  float rinv = rsqrtf(var + 1e-5f);
  _Float16* orow = out + (size_t)row * DIM;
#pragma unroll
  for (int i = 0; i < 3; i++) {
    int c = tid + i * TB;
    orow[c] = (_Float16)((v[i] - mean) * rinv * gg[c] + bb[c]);
  }
}

// ---------------- generic LDS-tiled WMMA GEMM --------------------------------
// Block tile 128x64, 8 waves (4 in M x 2 in N), each wave a 32x32 tile:
// 2 A-frags x 2 B-frags -> 4 wmma per K-step (each fragment reused twice).
// B tile stored transposed in LDS so all fragment loads are ds_load_b128.
struct GemmP {
  const _Float16* A;     // amode 0: f16 row-major [M x K]
  int amode;             // 0: f16 A ; 1: A = c(+pos) f32 from d_out score maps
  int src_sel;           // amode1: 0=cr 1=ct
  int addpos;            // amode1: add pos_emb
  const float* pos;      // [NS x NN]
  const _Float16* Bw;    // f16 row-major [K x Ncol]
  const float* bias;     // nullable, len Ncol
  int M, Ncol, K;
  int m_dyn;             // 1 => M = BB*NKEEP
  int epi;               // 0 qkv-scatter, 1 f16, 2 f16+gelu, 3 f32+resid
  _Float16* out_h;
  float* out_f;
  const float* resid;
  int dout_sel;          // epi3: -1 none, 0/1 -> d_out x0/x1
  _Float16 *qh, *kh, *vh;
  const int* kr;
  float* dout;
};

__global__ void __launch_bounds__(TB) k_gemm(GemmP p) {
  Off off = mkoff(p.kr);
  int M = p.m_dyn ? BB * off.NKEEP : p.M;
  int bm = blockIdx.x * 128, bn = blockIdx.y * 64;
  if (bm >= M) return;
  __shared__ _Float16 As[128][32];
  __shared__ _Float16 Bt[64][32];   // transposed: [n][k]
  int tid = threadIdx.x, lane = tid & 31, w = tid >> 5;
  int wm = w >> 1, wn = w & 1;      // wave tile: rows wm*32, cols wn*32
  int koff = (lane >> 4) << 3, rr = lane & 15;
  v8f acc00 = {0.f, 0.f, 0.f, 0.f, 0.f, 0.f, 0.f, 0.f};
  v8f acc01 = acc00, acc10 = acc00, acc11 = acc00;
  const float* cbase = nullptr;
  float posw = p.addpos ? 1.f : 0.f;
  if (p.amode == 1) cbase = p.dout + (p.src_sel ? off.ct : off.cr);

  for (int k0 = 0; k0 < p.K; k0 += 32) {
    {   // A tile 128x32 : each thread fills 16 halfs of one row-half
      int r = tid >> 1, c = (tid & 1) * 16;
      int gr = bm + r; if (gr > M - 1) gr = M - 1;
      if (p.amode == 0) {
        const _Float16* s = p.A + (size_t)gr * p.K + k0 + c;
        *(v8h*)&As[r][c] = *(const v8h*)s;
        *(v8h*)&As[r][c + 8] = *(const v8h*)(s + 8);
      } else {
        int bh = gr / NS, qi = gr - bh * NS;
        const float* s = cbase + ((size_t)bh * NN + NT + qi) * NN + k0 + c;
        const float* pp = p.pos + (size_t)qi * NN + k0 + c;
#pragma unroll
        for (int i = 0; i < 16; i++)
          As[r][c + i] = (_Float16)(s[i] + posw * pp[i]);
      }
    }
    {   // B tile 32x64, stored transposed
      int r = tid >> 3, c = (tid & 7) * 8;
      const _Float16* s = p.Bw + (size_t)(k0 + r) * p.Ncol + bn + c;
#pragma unroll
      for (int i = 0; i < 8; i++) Bt[c + i][r] = s[i];
      if (k0 + 32 < p.K)
        __builtin_prefetch(p.Bw + (size_t)(k0 + 32 + r) * p.Ncol + bn + c, 0, 0);
    }
    __syncthreads();
    int am0 = wm * 32 + rr, am1 = am0 + 16;
    int ln0 = wn * 32 + rr, ln1 = ln0 + 16;
    v16h a0 = ldfrag_lds32(&As[am0][0], koff);
    v16h a1 = ldfrag_lds32(&As[am1][0], koff);
    v16h b0 = ldfrag_lds32(&Bt[ln0][0], koff);
    v16h b1 = ldfrag_lds32(&Bt[ln1][0], koff);
    acc00 = wmma16(a0, b0, acc00);
    acc01 = wmma16(a0, b1, acc01);
    acc10 = wmma16(a1, b0, acc10);
    acc11 = wmma16(a1, b1, acc11);
    __syncthreads();
  }

  float* outf = p.out_f;
  if (p.epi == 3 && p.dout_sel >= 0) outf = p.dout + (p.dout_sel ? off.x1 : off.x0);
#pragma unroll
  for (int mi = 0; mi < 2; mi++) {
#pragma unroll
    for (int g = 0; g < 8; g++) {
      int m = bm + wm * 32 + mi * 16 + g + ((lane >> 4) << 3);
      if (m >= M) continue;
#pragma unroll
      for (int u = 0; u < 2; u++) {
        int n = bn + wn * 32 + u * 16 + (lane & 15);
        float v = mi ? (u ? acc11[g] : acc10[g]) : (u ? acc01[g] : acc00[g]);
        if (p.bias) v += p.bias[n];
        if (p.epi == 0) {
          int part = n / DIM, rem = n - part * DIM;
          int h = rem >> 6, d = rem & 63;
          int b = m / NN, nn2 = m - b * NN;
          size_t di = (((size_t)b * HEADS + h) * NN + nn2) * HD + d;
          _Float16* dst = (part == 0) ? p.qh : (part == 1) ? p.kh : p.vh;
          dst[di] = (_Float16)v;
        } else if (p.epi == 1) {
          p.out_h[(size_t)m * p.Ncol + n] = (_Float16)v;
        } else if (p.epi == 2) {
          float gel = 0.5f * v * (1.f + erff(v * 0.70710678f));
          p.out_h[(size_t)m * p.Ncol + n] = (_Float16)gel;
        } else {
          outf[(size_t)m * p.Ncol + n] = v + p.resid[(size_t)m * p.Ncol + n];
        }
      }
    }
  }
}

// ---------------- attention scores: c = mask ? NEG : (Q K^T)*SCL -------------
__global__ void __launch_bounds__(TB) k_scores(const _Float16* qh, const _Float16* kh,
                                               const unsigned char* mask, int sel,
                                               const int* kr, float* dout) {
  Off off = mkoff(kr);
  float* c = dout + (sel ? off.ct : off.cr);
  int bh = blockIdx.z, b = bh / HEADS;
  const _Float16* q = qh + (size_t)bh * NN * HD;
  const _Float16* k = kh + (size_t)bh * NN * HD;
  int lane = threadIdx.x & 31, w = threadIdx.x >> 5;
  int wm = w >> 1, wn = w & 1;
  int bm = blockIdx.x * 64 + wm * 16, bn = blockIdx.y * 64 + wn * 32;
  v8f a0 = {0.f, 0.f, 0.f, 0.f, 0.f, 0.f, 0.f, 0.f};
  v8f a1 = a0;
  for (int k0 = 0; k0 < HD; k0 += 32) {
    v16h af = ldfrag_rm(q, bm, k0, HD);
    v16h b0 = ldfrag_rm(k, bn, k0, HD);
    v16h b1 = ldfrag_rm(k, bn + 16, k0, HD);
    a0 = wmma16(af, b0, a0);
    a1 = wmma16(af, b1, a1);
  }
  size_t rowb = (size_t)bh * NN;
#pragma unroll
  for (int g = 0; g < 8; g++) {
    int m = bm + g + ((lane >> 4) << 3);
#pragma unroll
    for (int u = 0; u < 2; u++) {
      int n = bn + u * 16 + (lane & 15);
      float v = mask[b * NN + n] ? NEGV : (u ? a1[g] : a0[g]) * SCL;
      c[(rowb + m) * NN + n] = v;
    }
  }
}

// ---------------- CMA attention: softmax((qa kb^T)*MSCL) -> f16 --------------
__global__ void __launch_bounds__(TB) k_cma_attn(const _Float16* qsrc,
                                                 const _Float16* ksrc,
                                                 _Float16* out) {
  __shared__ float sm[16][NS];
  __shared__ float pr[16][16];
  int bh = blockIdx.y, bm = blockIdx.x * 16;
  const _Float16* q = qsrc + (size_t)bh * NS * MATCH;
  const _Float16* k = ksrc + (size_t)bh * NS * MATCH;
  int tid = threadIdx.x, lane = tid & 31, w = tid >> 5;
  v8f a0 = {0.f, 0.f, 0.f, 0.f, 0.f, 0.f, 0.f, 0.f};
  v8f a1 = a0;
  for (int k0 = 0; k0 < MATCH; k0 += 32) {
    v16h af = ldfrag_rm(q, bm, k0, MATCH);
    v16h b0 = ldfrag_rm(k, w * 32, k0, MATCH);
    v16h b1 = ldfrag_rm(k, w * 32 + 16, k0, MATCH);
    a0 = wmma16(af, b0, a0);
    a1 = wmma16(af, b1, a1);
  }
#pragma unroll
  for (int g = 0; g < 8; g++) {
    int m = g + ((lane >> 4) << 3);
    sm[m][w * 32 + (lane & 15)] = a0[g] * MSCL;
    sm[m][w * 32 + 16 + (lane & 15)] = a1[g] * MSCL;
  }
  __syncthreads();
  int row = tid >> 4, c0 = tid & 15;
  float mx = -1e30f;
  for (int c = c0; c < NS; c += 16) mx = fmaxf(mx, sm[row][c]);
  pr[row][c0] = mx; __syncthreads();
  if (c0 == 0) { float m2 = pr[row][0]; for (int i = 1; i < 16; i++) m2 = fmaxf(m2, pr[row][i]); pr[row][0] = m2; }
  __syncthreads();
  mx = pr[row][0]; __syncthreads();
  float s = 0.f;
  for (int c = c0; c < NS; c += 16) s += __expf(sm[row][c] - mx);
  pr[row][c0] = s; __syncthreads();
  if (c0 == 0) { float t = 0.f; for (int i = 0; i < 16; i++) t += pr[row][i]; pr[row][0] = t; }
  __syncthreads();
  float inv = 1.f / pr[row][0];
  _Float16* o = out + ((size_t)bh * NS + bm + row) * NS;
  for (int c = c0; c < NS; c += 16) o[c] = (_Float16)(__expf(sm[row][c] - mx) * inv);
}

// ---------------- mod = a @ v ; c[..., NT:, :NT] += mod ----------------------
__global__ void __launch_bounds__(TB) k_mod(const _Float16* a, const _Float16* v,
                                            int sel, const int* kr, float* dout) {
  Off off = mkoff(kr);
  float* c = dout + (sel ? off.ct : off.cr);
  __shared__ _Float16 Vt[64][32];   // transposed V tile [n][k]
  int tid = threadIdx.x, lane = tid & 31, w = tid >> 5;
  int wm = w >> 1, wn = w & 1;
  int bm = blockIdx.x * 64 + wm * 16;
  int bh = (blockIdx.x * 64) / NS;
  const _Float16* vb = v + (size_t)bh * NS * VDIM;
  int koff = (lane >> 4) << 3, rr = lane & 15;
  v8f a0 = {0.f, 0.f, 0.f, 0.f, 0.f, 0.f, 0.f, 0.f};
  v8f a1 = a0;
  int ln0 = wn * 32 + rr, ln1 = ln0 + 16;
  for (int k0 = 0; k0 < NS; k0 += 32) {
    {   // stage V tile 32x64 transposed
      int r = tid >> 3, cc = (tid & 7) * 8;
      const _Float16* s = vb + (size_t)(k0 + r) * VDIM + cc;
#pragma unroll
      for (int i = 0; i < 8; i++) Vt[cc + i][r] = s[i];
    }
    __syncthreads();
    v16h af = ldfrag_rm(a, bm, k0, NS);
    v16h b0 = ldfrag_lds32(&Vt[ln0][0], koff);
    v16h b1 = ldfrag_lds32(&Vt[ln1][0], koff);
    a0 = wmma16(af, b0, a0);
    a1 = wmma16(af, b1, a1);
    __syncthreads();
  }
#pragma unroll
  for (int g = 0; g < 8; g++) {
    int gm = bm + g + ((lane >> 4) << 3);
    int ql = gm - bh * NS;
#pragma unroll
    for (int u = 0; u < 2; u++) {
      int n = wn * 32 + u * 16 + (lane & 15);
      size_t ci = ((size_t)bh * NN + NT + ql) * NN + n;
      c[ci] += (u ? a1[g] : a0[g]);
    }
  }
}

// ---------------- full-row softmax over cr then ct (contiguous) --------------
__global__ void __launch_bounds__(TB) k_softmax(const int* kr, float* dout) {
  Off off = mkoff(kr);
  float* row = dout + off.cr + (size_t)blockIdx.x * NN;
  __shared__ float sh[TB];
  int tid = threadIdx.x;
  float v0 = row[tid];
  float v1 = (tid < NN - TB) ? row[TB + tid] : -1e30f;
  sh[tid] = fmaxf(v0, v1); __syncthreads();
  for (int st = 128; st > 0; st >>= 1) { if (tid < st) sh[tid] = fmaxf(sh[tid], sh[tid + st]); __syncthreads(); }
  float mx = sh[0]; __syncthreads();
  float e0 = __expf(v0 - mx), e1 = (tid < NN - TB) ? __expf(v1 - mx) : 0.f;
  sh[tid] = e0 + e1; __syncthreads();
  for (int st = 128; st > 0; st >>= 1) { if (tid < st) sh[tid] += sh[tid + st]; __syncthreads(); }
  float inv = 1.f / sh[0];
  row[tid] = e0 * inv;
  if (tid < NN - TB) row[TB + tid] = e1 * inv;
}

// ---------------- attn_t[b][s] = mean_{h,q<NT}(0.5 cr + 0.5 ct) --------------
__global__ void __launch_bounds__(TB) k_attn_t(const int* kr, const float* dout,
                                               float* attn_t) {
  Off off = mkoff(kr);
  int b = blockIdx.x, s = threadIdx.x;
  const float* cr = dout + off.cr + (size_t)b * HEADS * NN * NN;
  const float* ct = dout + off.ct + (size_t)b * HEADS * NN * NN;
  float acc = 0.f;
  for (int h = 0; h < HEADS; h++)
    for (int q = 0; q < NT; q++) {
      size_t i = ((size_t)h * NN + q) * NN + NT + s;
      acc += cr[i] + ct[i];
    }
  attn_t[b * NS + s] = 0.5f * acc / (float)(HEADS * NT);
}

// ---------------- top-k (bitonic, stable like argsort(-x)) -------------------
__global__ void __launch_bounds__(TB) k_topk(const float* attn_t, const int* gis,
                                             const int* git, int* order,
                                             const int* kr, float* dout) {
  Off off = mkoff(kr);
  int b = blockIdx.x, tid = threadIdx.x;
  __shared__ float sv[NS];
  __shared__ int si[NS];
  sv[tid] = attn_t[b * NS + tid]; si[tid] = tid; __syncthreads();
  for (int k = 2; k <= NS; k <<= 1)
    for (int j = k >> 1; j > 0; j >>= 1) {
      int x = tid ^ j;
      if (x > tid) {
        float v1 = sv[tid], v2 = sv[x];
        int i1 = si[tid], i2 = si[x];
        bool first = (v1 > v2) || (v1 == v2 && i1 < i2);
        bool up = ((tid & k) == 0);
        if (up ? !first : first) { sv[tid] = v2; sv[x] = v1; si[tid] = i2; si[x] = i1; }
      }
      __syncthreads();
    }
  order[b * NS + tid] = si[tid];
  if (tid < off.LK)
    dout[off.ki + (size_t)b * off.LK + tid] = (float)gis[b * NS + si[tid]];
  else
    dout[off.ri + (size_t)b * (NS - off.LK) + (tid - off.LK)] = (float)gis[b * NS + si[tid]];
  if (tid < NT) dout[off.gt + (size_t)b * NT + tid] = (float)git[b * NT + tid];
}

// ---------------- o = attn @ v  (per head), o layout [B,N,DIM] f16 -----------
__global__ void __launch_bounds__(TB) k_attn_out(const _Float16* vh, int sel,
                                                 const int* kr, const float* dout,
                                                 _Float16* oh) {
  Off off = mkoff(kr);
  const float* cb = dout + (sel ? off.ct : off.cr);
  int bh = blockIdx.y, b = bh / HEADS, h = bh - b * HEADS;
  const float* attn = cb + (size_t)bh * NN * NN;
  const _Float16* v = vh + (size_t)bh * NN * HD;
  __shared__ _Float16 As[32][32];
  __shared__ _Float16 Vt[64][32];   // transposed V tile [n][k]
  int tid = threadIdx.x, lane = tid & 31, w = tid >> 5;
  int wm = w >> 2, wn = w & 3;
  int bm = blockIdx.x * 32;
  int koff = (lane >> 4) << 3, rr = lane & 15;
  v8f acc = {0.f, 0.f, 0.f, 0.f, 0.f, 0.f, 0.f, 0.f};
  for (int k0 = 0; k0 < NN; k0 += 32) {
    {   // stage attn 32x32 (f32 -> f16)
      int r = tid >> 3, cc = (tid & 7) * 4;
      const float* s = attn + (size_t)(bm + r) * NN + k0 + cc;
#pragma unroll
      for (int i = 0; i < 4; i++) As[r][cc + i] = (_Float16)s[i];
    }
    {   // stage V 32x64 transposed
      int r = tid >> 3, cc = (tid & 7) * 8;
      const _Float16* s = v + (size_t)(k0 + r) * HD + cc;
#pragma unroll
      for (int i = 0; i < 8; i++) Vt[cc + i][r] = s[i];
    }
    __syncthreads();
    int am = wm * 16 + rr, n = wn * 16 + rr;
    v16h af = ldfrag_lds32(&As[am][0], koff);
    v16h bf = ldfrag_lds32(&Vt[n][0], koff);
    acc = wmma16(af, bf, acc);
    __syncthreads();
  }
#pragma unroll
  for (int g = 0; g < 8; g++) {
    int m = bm + wm * 16 + g + ((lane >> 4) << 3);
    int n = wn * 16 + rr;
    oh[((size_t)b * NN + m) * DIM + h * HD + n] = (_Float16)acc[g];
  }
}

// ---------------- gather pruned tokens (compact rows) ------------------------
__global__ void __launch_bounds__(TB) k_prune(const float* xatt, const int* order,
                                              const int* kr, float* xpr) {
  Off off = mkoff(kr);
  int blk = blockIdx.x, b = blk / NN, t = blk - b * NN;
  if (t >= off.NKEEP) return;
  int src = (t < NT) ? t : NT + order[b * NS + (t - NT)];
  const float* s = xatt + ((size_t)b * NN + src) * DIM;
  float* d = xpr + ((size_t)b * off.NKEEP + t) * DIM;
  for (int i = threadIdx.x; i < DIM; i += TB) d[i] = s[i];
}

// =============================================================================
extern "C" void kernel_launch(void* const* d_in, const int* in_sizes, int n_in,
                              void* d_out, int out_size, void* d_ws, size_t ws_size,
                              hipStream_t stream) {
  (void)in_sizes; (void)n_in; (void)out_size; (void)ws_size;
  const float* x_rgb = (const float*)d_in[0];
  const float* x_tir = (const float*)d_in[1];
  const int* git = (const int*)d_in[2];
  const int* gis = (const int*)d_in[3];
  const unsigned char* mask = (const unsigned char*)d_in[4];
  const int* kr = (const int*)d_in[6];
  const float* pos = (const float*)d_in[7];
  const float* n1g = (const float*)d_in[9];
  const float* n1b = (const float*)d_in[10];
  const float* qkv_w = (const float*)d_in[11];
  const float* proj_w = (const float*)d_in[12];
  const float* proj_b = (const float*)d_in[13];
  const float* stq_w = (const float*)d_in[14];
  const float* stq_b = (const float*)d_in[15];
  const float* stk_w = (const float*)d_in[16];
  const float* stk_b = (const float*)d_in[17];
  const float* stv_w = (const float*)d_in[18];
  const float* stv_b = (const float*)d_in[19];
  const float* n2g = (const float*)d_in[20];
  const float* n2b = (const float*)d_in[21];
  const float* fc1_w = (const float*)d_in[22];
  const float* fc1_b = (const float*)d_in[23];
  const float* fc2_w = (const float*)d_in[24];
  const float* fc2_b = (const float*)d_in[25];
  float* dout = (float*)d_out;

  char* ws = (char*)d_ws;
  size_t o = 0;
  auto alloc = [&](size_t bytes) { char* p = ws + o; o += (bytes + 255) & ~(size_t)255; return p; };
  const size_t XH = (size_t)BB * NN * DIM * 2;               // 15.7MB f16
  _Float16* wqkv = (_Float16*)alloc((size_t)DIM * 3 * DIM * 2);
  _Float16* wproj = (_Float16*)alloc((size_t)DIM * DIM * 2);
  _Float16* wfc1 = (_Float16*)alloc((size_t)DIM * MLPH * 2);
  _Float16* wfc2 = (_Float16*)alloc((size_t)MLPH * DIM * 2);
  _Float16* wstq = (_Float16*)alloc((size_t)NN * MATCH * 2);
  _Float16* wstk = (_Float16*)alloc((size_t)NN * MATCH * 2);
  _Float16* wstv = (_Float16*)alloc((size_t)NN * VDIM * 2);
  _Float16* xnh0 = (_Float16*)alloc(XH);
  _Float16* xnh1 = (_Float16*)alloc(XH);
  _Float16* qh0 = (_Float16*)alloc(XH);
  _Float16* kh0 = (_Float16*)alloc(XH);
  _Float16* vh0 = (_Float16*)alloc(XH);
  _Float16* qh1 = (_Float16*)alloc(XH);
  _Float16* kh1 = (_Float16*)alloc(XH);
  _Float16* vh1 = (_Float16*)alloc(XH);
  const size_t CQ = (size_t)BB * HEADS * NS * MATCH * 2;
  _Float16* cq0 = (_Float16*)alloc(CQ);
  _Float16* ck0 = (_Float16*)alloc(CQ);
  _Float16* cv0 = (_Float16*)alloc(CQ);
  _Float16* cq1 = (_Float16*)alloc(CQ);
  _Float16* ck1 = (_Float16*)alloc(CQ);
  _Float16* cv1 = (_Float16*)alloc(CQ);
  const size_t ASZ = (size_t)BB * HEADS * NS * NS * 2;
  _Float16* aab = (_Float16*)alloc(ASZ);
  _Float16* aba = (_Float16*)alloc(ASZ);
  _Float16* oh = (_Float16*)alloc(XH);
  float* xatt0 = (float*)alloc((size_t)BB * NN * DIM * 4);
  float* xatt1 = (float*)alloc((size_t)BB * NN * DIM * 4);
  float* xpr0 = (float*)alloc((size_t)BB * NN * DIM * 4);
  float* xpr1 = (float*)alloc((size_t)BB * NN * DIM * 4);
  _Float16* xpnh = (_Float16*)alloc(XH);
  _Float16* hbuf = (_Float16*)alloc((size_t)BB * NN * MLPH * 2);
  float* attn_t = (float*)alloc((size_t)BB * NS * 4);
  int* order = (int*)alloc((size_t)BB * NS * 4);

  // weight conversions
  auto cvt = [&](const float* s, _Float16* d, int n) {
    k_f2h<<<(n + TB - 1) / TB, TB, 0, stream>>>(s, d, n);
  };
  cvt(qkv_w, wqkv, DIM * 3 * DIM);
  cvt(proj_w, wproj, DIM * DIM);
  cvt(fc1_w, wfc1, DIM * MLPH);
  cvt(fc2_w, wfc2, MLPH * DIM);
  cvt(stq_w, wstq, NN * MATCH);
  cvt(stk_w, wstk, NN * MATCH);
  cvt(stv_w, wstv, NN * VDIM);

  // layernorm 1
  k_ln<<<BB * NN, TB, 0, stream>>>(x_rgb, n1g, n1b, xnh0, 0, kr);
  k_ln<<<BB * NN, TB, 0, stream>>>(x_tir, n1g, n1b, xnh1, 0, kr);

  // QKV GEMMs (M=10240 -> 80 blocks of 128)
  GemmP p{};
  p.kr = kr; p.dout = dout; p.amode = 0; p.Bw = wqkv; p.bias = nullptr;
  p.M = BB * NN; p.Ncol = 3 * DIM; p.K = DIM; p.epi = 0; p.dout_sel = -1;
  p.A = xnh0; p.qh = qh0; p.kh = kh0; p.vh = vh0;
  k_gemm<<<dim3(80, 36), TB, 0, stream>>>(p);
  p.A = xnh1; p.qh = qh1; p.kh = kh1; p.vh = vh1;
  k_gemm<<<dim3(80, 36), TB, 0, stream>>>(p);

  // attention scores -> d_out cr/ct
  k_scores<<<dim3(5, 5, BB * HEADS), TB, 0, stream>>>(qh0, kh0, mask, 0, kr, dout);
  k_scores<<<dim3(5, 5, BB * HEADS), TB, 0, stream>>>(qh1, kh1, mask, 1, kr, dout);

  // CMA projections (A = c(+pos) f32 from d_out; M=98304 -> 768 blocks)
  GemmP q{};
  q.kr = kr; q.dout = dout; q.amode = 1; q.pos = pos;
  q.M = BB * HEADS * NS; q.Ncol = MATCH; q.K = NN; q.epi = 1; q.dout_sel = -1;
  q.src_sel = 0; q.addpos = 1; q.Bw = wstq; q.bias = stq_b; q.out_h = cq0;
  k_gemm<<<dim3(768, 1), TB, 0, stream>>>(q);
  q.Bw = wstk; q.bias = stk_b; q.out_h = ck0;
  k_gemm<<<dim3(768, 1), TB, 0, stream>>>(q);
  q.addpos = 0; q.Bw = wstv; q.bias = stv_b; q.out_h = cv0;
  k_gemm<<<dim3(768, 1), TB, 0, stream>>>(q);
  q.src_sel = 1; q.addpos = 1; q.Bw = wstq; q.bias = stq_b; q.out_h = cq1;
  k_gemm<<<dim3(768, 1), TB, 0, stream>>>(q);
  q.Bw = wstk; q.bias = stk_b; q.out_h = ck1;
  k_gemm<<<dim3(768, 1), TB, 0, stream>>>(q);
  q.addpos = 0; q.Bw = wstv; q.bias = stv_b; q.out_h = cv1;
  k_gemm<<<dim3(768, 1), TB, 0, stream>>>(q);

  // CMA attention maps + mod add
  k_cma_attn<<<dim3(16, BB * HEADS), TB, 0, stream>>>(cq0, ck1, aab);
  k_cma_attn<<<dim3(16, BB * HEADS), TB, 0, stream>>>(cq1, ck0, aba);
  k_mod<<<dim3(1536), TB, 0, stream>>>(aab, cv1, 0, kr, dout);
  k_mod<<<dim3(1536), TB, 0, stream>>>(aba, cv0, 1, kr, dout);

  // softmax over all rows of cr and ct (contiguous in d_out)
  k_softmax<<<dim3(2 * BB * HEADS * NN), TB, 0, stream>>>(kr, dout);

  // token scoring + top-k + index outputs
  k_attn_t<<<BB, TB, 0, stream>>>(kr, dout, attn_t);
  k_topk<<<BB, NS, 0, stream>>>(attn_t, gis, git, order, kr, dout);

  // attention output + projection (+residual) per stream
  GemmP r{};
  r.kr = kr; r.dout = dout; r.amode = 0; r.Bw = wproj; r.bias = proj_b;
  r.M = BB * NN; r.Ncol = DIM; r.K = DIM; r.epi = 3; r.dout_sel = -1;
  k_attn_out<<<dim3(10, BB * HEADS), TB, 0, stream>>>(vh0, 0, kr, dout, oh);
  r.A = oh; r.resid = x_rgb; r.out_f = xatt0;
  k_gemm<<<dim3(80, 12), TB, 0, stream>>>(r);
  k_attn_out<<<dim3(10, BB * HEADS), TB, 0, stream>>>(vh1, 1, kr, dout, oh);
  r.A = oh; r.resid = x_tir; r.out_f = xatt1;
  k_gemm<<<dim3(80, 12), TB, 0, stream>>>(r);

  // prune
  k_prune<<<BB * NN, TB, 0, stream>>>(xatt0, order, kr, xpr0);
  k_prune<<<BB * NN, TB, 0, stream>>>(xatt1, order, kr, xpr1);

  // MLP per stream (dynamic M = BB*NKEEP, grid covers max M=10240)
  for (int s = 0; s < 2; s++) {
    const float* xp = s ? xpr1 : xpr0;
    k_ln<<<BB * NN, TB, 0, stream>>>(xp, n2g, n2b, xpnh, 1, kr);
    GemmP f1{};
    f1.kr = kr; f1.dout = dout; f1.amode = 0; f1.A = xpnh; f1.Bw = wfc1;
    f1.bias = fc1_b; f1.M = BB * NN; f1.Ncol = MLPH; f1.K = DIM; f1.m_dyn = 1;
    f1.epi = 2; f1.out_h = hbuf; f1.dout_sel = -1;
    k_gemm<<<dim3(80, 48), TB, 0, stream>>>(f1);
    GemmP f2{};
    f2.kr = kr; f2.dout = dout; f2.amode = 0; f2.A = hbuf; f2.Bw = wfc2;
    f2.bias = fc2_b; f2.M = BB * NN; f2.Ncol = DIM; f2.K = MLPH; f2.m_dyn = 1;
    f2.epi = 3; f2.resid = xp; f2.dout_sel = s;
    k_gemm<<<dim3(80, 12), TB, 0, stream>>>(f2);
  }
}
